// UnpackSlidingWindowAttention_26980984553604
// MI455X (gfx1250) — matrix operand order repaired and checked by reading, not compiled, not run on hardware
//
#include <hip/hip_runtime.h>

#define SEQ_LEN 4096
#define DIM 1024
#define HEADS 8
#define HD 128
#define BS 64
#define NB 64
#define PACK 64
#define PENV -10000.0f
#define RSQ 0.08838834764831845f   // 1/sqrt(128)

typedef __attribute__((ext_vector_type(16))) __bf16 v16bf;
typedef __attribute__((ext_vector_type(8)))  __bf16 v8bf;
typedef __attribute__((ext_vector_type(8)))  float  v8f;
typedef __attribute__((ext_vector_type(4)))  unsigned int v4u;
typedef __attribute__((ext_vector_type(8)))  int v8i;
typedef __attribute__((ext_vector_type(4)))  int v4i;

__device__ __forceinline__ __bf16 f2bf(float f) {
  unsigned u = __builtin_bit_cast(unsigned, f);
  unsigned r = (u + 0x7FFFu + ((u >> 16) & 1u)) >> 16;
  unsigned short s = (unsigned short)r;
  return __builtin_bit_cast(__bf16, s);
}

// ---- A fragment (16x32 bf16, element [m][k], row-major, leading dim ld)
// lanes 0-15 -> M=0..15, K=0..15 ; lanes 16-31 -> K=16..31 (contiguous, matches B)
__device__ __forceinline__ v16bf load_a(const __bf16* p, int ld) {
  int lane = threadIdx.x & 31;
  int m  = lane & 15;
  int kb = (lane >> 4) << 4;
  return *(const v16bf*)(p + m * ld + kb);
}

// ---- B fragment (32x16 bf16, element [k][n]) from [n][k] storage (ld along k).
// This code's order: lane n=l&15; lanes 0-15 hold K=0..15, lanes 16-31 hold K=16..31 (contiguous, same as A)
// -> one 32-byte contiguous LDS load per fragment.
__device__ __forceinline__ v16bf load_b_nk(const __bf16* p, int ld) {
  int lane = threadIdx.x & 31;
  int n  = lane & 15;
  int kb = (lane >> 4) << 4;          // 0 or 16
  return *(const v16bf*)(p + n * ld + kb);
}

// ---- Tensor Data Mover: 2-D tile (bf16) global -> LDS, issued once per wave.
// D# per CDNA5 ISA 8.3/8.4. Descriptor groups 2/3 zero (<=2D tensor).
__device__ __forceinline__ void tdm_load_2d_bf16(void* lds, const void* gsrc,
                                                 unsigned tile_d0, unsigned tile_d1,
                                                 unsigned tensor_d0, unsigned tensor_d1,
                                                 unsigned stride_d0) {
#if __has_builtin(__builtin_amdgcn_tensor_load_to_lds)
  unsigned long long ga = (unsigned long long)gsrc;
  unsigned lds_off = (unsigned)(unsigned long long)lds;   // generic ptr low 32 = LDS offset
  v4u g0;
  g0[0] = 1u;                                             // count=1, user D#
  g0[1] = lds_off;                                        // lds_addr
  g0[2] = (unsigned)(ga & 0xFFFFFFFFu);                   // global_addr[31:0]
  g0[3] = (unsigned)((ga >> 32) & 0x01FFFFFFu) | (2u << 30); // addr[56:32] | type=2
  v8i g1;
  g1[0] = (int)(1u << 16);                                // data_size=1 (2 bytes)
  g1[1] = (int)(tensor_d0 << 16);                         // tensor_dim0[15:0]
  g1[2] = (int)((tensor_d0 >> 16) | (tensor_d1 << 16));   // dim0 hi | dim1 lo
  g1[3] = (int)((tensor_d1 >> 16) | (tile_d0 << 16));     // dim1 hi | tile_dim0
  g1[4] = (int)(tile_d1 & 0xFFFFu);                       // tile_dim1 (tile_dim2=0)
  g1[5] = (int)stride_d0;                                 // tensor_dim0_stride[31:0]
  g1[6] = 0;                                              // stride hi | dim1_stride lo
  g1[7] = 0;
  v4i z4 = {0, 0, 0, 0};
#if defined(__clang_major__) && __clang_major__ >= 23
  v8i z8 = {0, 0, 0, 0, 0, 0, 0, 0};
  __builtin_amdgcn_tensor_load_to_lds(g0, g1, z4, z4, z8, 0);
#else
  __builtin_amdgcn_tensor_load_to_lds(g0, g1, z4, z4, 0);
#endif
  __builtin_amdgcn_s_wait_tensorcnt(0);
#else
  // fallback: cooperative vector copy by wave 0 (32 lanes, 64x128 tile)
  int lane = threadIdx.x & 31;
  for (int r = lane; r < (int)tile_d1; r += 32) {
    const uint4* s = (const uint4*)((const __bf16*)gsrc + (size_t)r * stride_d0);
    uint4* d = (uint4*)((__bf16*)lds + (size_t)r * tile_d0);
    for (unsigned c = 0; c < tile_d0 / 8; ++c) d[c] = s[c];
  }
#endif
}

// ---------------------------------------------------------------- convert
__global__ void __launch_bounds__(256) f32_to_bf16_kernel(const float* __restrict__ in,
                                                          __bf16* __restrict__ out, int n) {
  int i = blockIdx.x * blockDim.x + threadIdx.x;
  if (i < n) out[i] = f2bf(in[i]);
}

// ---------------------------------------------------------------- GEMM: C = bf16(A*W + bias)
// A [M,K] bf16, W [K,N] bf16, C [M,N] bf16. 64x64 tile/block, 8 waves, W staged transposed.
__global__ void __launch_bounds__(256) gemm_bias_bf16(const __bf16* __restrict__ A,
                                                      const __bf16* __restrict__ W,
                                                      const float* __restrict__ bias,
                                                      __bf16* __restrict__ C,
                                                      int M, int N, int K) {
  __shared__ __bf16 sA[64 * 32];     // [m][k]
  __shared__ __bf16 sBt[64 * 32];    // [n][k]  (transposed W tile)
  const int t = threadIdx.x;
  const int m0 = blockIdx.y * 64;
  const int n0 = blockIdx.x * 64;
  const int wave = t >> 5;
  const int mt  = wave >> 1;
  const int ntb = (wave & 1) * 2;
  v8f acc0 = {}; v8f acc1 = {};

  const int ar = t >> 2, ac = (t & 3) * 8;   // A staging: 64 rows x 32 cols
  const int br = t >> 3, bc = (t & 7) * 8;   // W staging: 32 k-rows x 64 n-cols

  for (int k0 = 0; k0 < K; k0 += 32) {
    *(uint4*)(sA + ar * 32 + ac) = *(const uint4*)(A + (size_t)(m0 + ar) * K + k0 + ac);
    v8bf wv = *(const v8bf*)(W + (size_t)(k0 + br) * N + n0 + bc);
#pragma unroll
    for (int e = 0; e < 8; ++e) sBt[(bc + e) * 32 + br] = wv[e];   // transpose into LDS
    if (k0 + 32 < K) {                                             // -> global_prefetch_b8
      __builtin_prefetch(A + (size_t)(m0 + ar) * K + k0 + 32 + ac, 0, 1);
      __builtin_prefetch(W + (size_t)(k0 + 32 + br) * N + n0 + bc, 0, 1);
    }
    __syncthreads();
    v16bf a  = load_a(sA + mt * 16 * 32, 32);
    v16bf b0 = load_b_nk(sBt + (ntb * 16) * 32, 32);
    v16bf b1 = load_b_nk(sBt + ((ntb + 1) * 16) * 32, 32);
    acc0 = __builtin_amdgcn_wmma_f32_16x16x32_bf16(false, a, false, b0, (short)0, acc0, false, false);
    acc1 = __builtin_amdgcn_wmma_f32_16x16x32_bf16(false, a, false, b1, (short)0, acc1, false, false);
    __syncthreads();
  }
  const int lane = t & 31;
  const int mo = (lane >> 4) * 8;
  const int nn = lane & 15;
  int ng0 = n0 + ntb * 16 + nn;
  int ng1 = n0 + (ntb + 1) * 16 + nn;
  float bv0 = bias[ng0], bv1 = bias[ng1];
#pragma unroll
  for (int r = 0; r < 8; ++r) {
    int row = m0 + mt * 16 + mo + r;
    C[(size_t)row * N + ng0] = f2bf(acc0[r] + bv0);
    C[(size_t)row * N + ng1] = f2bf(acc1[r] + bv1);
  }
}

// ---------------------------------------------------------------- attention
__device__ __forceinline__ int chunk_block(int qb, int c) {
  if (qb < 2) return c;                                   // first rows: blocks 0..3
  if (qb == NB - 1) return (c == 0) ? 0 : (NB - 4 + c);   // last: 0,61,62,63
  return (c == 0) ? 0 : (qb - 2 + c);                     // middle: 0, qb-1, qb, qb+1
}

// V chunk staged transposed: dst[n=128][k=64]
__device__ __forceinline__ void stage_v_transposed(__bf16* dst, const __bf16* src, int src_ld) {
  int t = threadIdx.x;
  int r  = t >> 2;            // k row 0..63
  int c0 = (t & 3) * 32;      // n col base
  const v8bf* s = (const v8bf*)(src + (size_t)r * src_ld + c0);
#pragma unroll
  for (int v = 0; v < 4; ++v) {
    v8bf g = s[v];
#pragma unroll
    for (int e = 0; e < 8; ++e) dst[(size_t)(c0 + v * 8 + e) * 64 + r] = g[e];
  }
}

#define ATTN_SMEM ((64 * 128 + 64 * 128 + 64 * 320) * 2 + 64 * 320 * 4)

__global__ void __launch_bounds__(256) attn_kernel(const __bf16* __restrict__ Q,
                                                   const __bf16* __restrict__ Kb,
                                                   const __bf16* __restrict__ Vb,
                                                   const __bf16* __restrict__ Kp,
                                                   const __bf16* __restrict__ Vp,
                                                   const float* __restrict__ mask,
                                                   float* __restrict__ out) {
  extern __shared__ __align__(16) char smem[];
  __bf16* sQ  = (__bf16*)smem;              // 64 x 128 [q][d]
  __bf16* sKV = sQ + 64 * 128;              // K chunk [j][d] / V chunk transposed [n][k]
  __bf16* sP  = sKV + 64 * 128;             // 64 x 320 probabilities (bf16)
  float*  sS  = (float*)(sP + 64 * 320);    // 64 x 320 scores (f32)

  const int qb = blockIdx.x;
  const int h  = blockIdx.y;
  const int t  = threadIdx.x;
  const int wave = t >> 5;
  const int lane = t & 31;
  const int mo = (lane >> 4) * 8;
  const int nn = lane & 15;
  const float slope = exp2f(-(float)(h + 1));   // 0.5^(h+1)
  const bool is_first = (qb < 2);
  const bool is_last  = (qb == NB - 1);
  const bool is_mid   = !is_first && !is_last;

  // Q tile via Tensor Data Mover (wave 0 issues; barrier publishes)
  if (wave == 0)
    tdm_load_2d_bf16(sQ, Q + (size_t)(qb * BS) * DIM + h * HD, HD, BS, DIM, SEQ_LEN, DIM);

  // ---- score pass: 5 chunks of 64 keys (4 band + 1 packed)
  const int mt  = wave >> 1;        // 0..3
  const int ntp = (wave & 1) * 2;   // 0 or 2
  for (int c = 0; c < 5; ++c) {
    __syncthreads();                // sKV free for reuse (also publishes sQ on c==0)
    int blk = (c == 4) ? 0 : chunk_block(qb, c);
    const __bf16* ksrc = (c == 4) ? (Kp + h * HD)
                                  : (Kb + (size_t)(blk * BS) * DIM + h * HD);
    if (wave == 0) tdm_load_2d_bf16(sKV, ksrc, HD, BS, DIM, SEQ_LEN, DIM);
    __syncthreads();
#pragma unroll
    for (int ni = 0; ni < 2; ++ni) {
      int nt = ntp + ni;
      v8f acc = {};
#pragma unroll
      for (int kk = 0; kk < HD; kk += 32) {
        v16bf a = load_a(sQ + mt * 16 * HD + kk, HD);
        v16bf b = load_b_nk(sKV + (nt * 16) * HD + kk, HD);  // K rows == [n][k] storage
        acc = __builtin_amdgcn_wmma_f32_16x16x32_bf16(false, a, false, b, (short)0, acc, false, false);
      }
      int jl = nt * 16 + nn;
#pragma unroll
      for (int r = 0; r < 8; ++r) {
        int i_loc = mt * 16 + mo + r;
        int i_abs = qb * BS + i_loc;
        float mq = mask[i_abs];
        float sc = acc[r] * RSQ;
        float pen;
        if (c == 4) {                          // packed path: Dp = slope*BS
          sc -= slope * (float)BS;
          pen = (1.f - mq) * PENV;
        } else {
          int j_real = blk * BS + jl;
          int j_bias = is_last ? ((NB - 4 + c) * BS + jl) : j_real;  // ref D[:,-BS:,-4BS:] quirk
          sc -= slope * fabsf((float)(i_abs - j_bias));
          if (is_last)              pen = 0.f;
          else if (is_mid && c > 0) pen = (1.f - mq * mask[j_real]) * PENV;
          else                      pen = (1.f - mq) * PENV;
        }
        sS[i_loc * 320 + c * 64 + jl] = sc + pen;
      }
    }
  }
  __syncthreads();

  // ---- dual softmax (band 0..255, packed 256..319); packed row scaled by mask_v
  if (t < 64) {
    float* row = sS + t * 320;
    __bf16* prow = sP + t * 320;
    float mx = -1e30f;
    for (int j = 0; j < 256; ++j) mx = fmaxf(mx, row[j]);
    float sum = 0.f;
    for (int j = 0; j < 256; ++j) { float e = __expf(row[j] - mx); row[j] = e; sum += e; }
    float inv = 1.f / sum;
    for (int j = 0; j < 256; ++j) prow[j] = f2bf(row[j] * inv);
    float mq = mask[qb * BS + t];
    mx = -1e30f;
    for (int j = 256; j < 320; ++j) mx = fmaxf(mx, row[j]);
    sum = 0.f;
    for (int j = 256; j < 320; ++j) { float e = __expf(row[j] - mx); row[j] = e; sum += e; }
    inv = mq / sum;
    for (int j = 256; j < 320; ++j) prow[j] = f2bf(row[j] * inv);
  }

  // ---- PV pass: O[64x128] = P[64x320] * V[320x128], V staged transposed [n][k]
  const int ntq = (wave & 1) * 4;
  v8f accs[4] = {};
  for (int c = 0; c < 5; ++c) {
    __syncthreads();
    int blk = (c == 4) ? 0 : chunk_block(qb, c);
    const __bf16* vsrc = (c == 4) ? (Vp + h * HD)
                                  : (Vb + (size_t)(blk * BS) * DIM + h * HD);
    stage_v_transposed(sKV, vsrc, DIM);
    __syncthreads();
#pragma unroll
    for (int kk = 0; kk < 64; kk += 32) {
      v16bf a = load_a(sP + mt * 16 * 320 + c * 64 + kk, 320);
#pragma unroll
      for (int j = 0; j < 4; ++j) {
        v16bf b = load_b_nk(sKV + ((ntq + j) * 16) * 64 + kk, 64);
        accs[j] = __builtin_amdgcn_wmma_f32_16x16x32_bf16(false, a, false, b, (short)0, accs[j], false, false);
      }
    }
  }

  // ---- epilogue with reference's raw-view index scramble:
  // out[h*512 + (s>>3)][(s&7)*128 + d] = val * mask[new_row]
#pragma unroll
  for (int j = 0; j < 4; ++j) {
    int d = (ntq + j) * 16 + nn;
#pragma unroll
    for (int r = 0; r < 8; ++r) {
      int i_abs = qb * BS + mt * 16 + mo + r;
      int row_new = h * 512 + (i_abs >> 3);
      int col_new = (i_abs & 7) * 128 + d;
      out[(size_t)row_new * DIM + col_new] = accs[j][r] * mask[row_new];
    }
  }
}

// ---------------------------------------------------------------- host
extern "C" void kernel_launch(void* const* d_in, const int* in_sizes, int n_in,
                              void* d_out, int out_size, void* d_ws, size_t ws_size,
                              hipStream_t stream) {
  (void)in_sizes; (void)n_in; (void)out_size; (void)ws_size;
  const float* X  = (const float*)d_in[0];
  const float* Cp = (const float*)d_in[1];
  const float* am = (const float*)d_in[2];
  const float* WQ = (const float*)d_in[3];
  const float* bQ = (const float*)d_in[4];
  const float* WK = (const float*)d_in[5];
  const float* bK = (const float*)d_in[6];
  const float* WV = (const float*)d_in[7];
  const float* bV = (const float*)d_in[8];
  float* out = (float*)d_out;

  char* w = (char*)d_ws;
  size_t off = 0;
  auto carve = [&](size_t bytes) { char* p = w + off; off += (bytes + 255) & ~(size_t)255; return p; };
  __bf16* Xbf  = (__bf16*)carve((size_t)SEQ_LEN * DIM * 2);
  __bf16* Cpbf = (__bf16*)carve((size_t)PACK * DIM * 2);
  __bf16* WQb  = (__bf16*)carve((size_t)DIM * DIM * 2);
  __bf16* WKb  = (__bf16*)carve((size_t)DIM * DIM * 2);
  __bf16* WVb  = (__bf16*)carve((size_t)DIM * DIM * 2);
  __bf16* Qb   = (__bf16*)carve((size_t)SEQ_LEN * DIM * 2);
  __bf16* Kbf  = (__bf16*)carve((size_t)SEQ_LEN * DIM * 2);
  __bf16* Vbf  = (__bf16*)carve((size_t)SEQ_LEN * DIM * 2);
  __bf16* Kp   = (__bf16*)carve((size_t)PACK * DIM * 2);
  __bf16* Vp   = (__bf16*)carve((size_t)PACK * DIM * 2);

  const int nX = SEQ_LEN * DIM, nW = DIM * DIM, nC = PACK * DIM;
  f32_to_bf16_kernel<<<(nX + 255) / 256, 256, 0, stream>>>(X,  Xbf,  nX);
  f32_to_bf16_kernel<<<(nC + 255) / 256, 256, 0, stream>>>(Cp, Cpbf, nC);
  f32_to_bf16_kernel<<<(nW + 255) / 256, 256, 0, stream>>>(WQ, WQb,  nW);
  f32_to_bf16_kernel<<<(nW + 255) / 256, 256, 0, stream>>>(WK, WKb,  nW);
  f32_to_bf16_kernel<<<(nW + 255) / 256, 256, 0, stream>>>(WV, WVb,  nW);

  dim3 gBig(DIM / 64, SEQ_LEN / 64), gSmall(DIM / 64, PACK / 64);
  gemm_bias_bf16<<<gBig,   256, 0, stream>>>(Xbf,  WQb, bQ, Qb,  SEQ_LEN, DIM, DIM);
  gemm_bias_bf16<<<gBig,   256, 0, stream>>>(Xbf,  WKb, bK, Kbf, SEQ_LEN, DIM, DIM);
  gemm_bias_bf16<<<gBig,   256, 0, stream>>>(Xbf,  WVb, bV, Vbf, SEQ_LEN, DIM, DIM);
  gemm_bias_bf16<<<gSmall, 256, 0, stream>>>(Cpbf, WKb, bK, Kp,  PACK,    DIM, DIM);
  gemm_bias_bf16<<<gSmall, 256, 0, stream>>>(Cpbf, WVb, bV, Vp,  PACK,    DIM, DIM);

  hipFuncSetAttribute(reinterpret_cast<const void*>(attn_kernel),
                      hipFuncAttributeMaxDynamicSharedMemorySize, ATTN_SMEM);
  attn_kernel<<<dim3(NB, HEADS), 256, ATTN_SMEM, stream>>>(Qb, Kbf, Vbf, Kp, Vp, am, out);
}
